// GATConvEncoderLayer_68264210202878
// MI455X (gfx1250) — compile-verified, matrix operand order checked by reading, not run on hardware
//
#include <hip/hip_runtime.h>
#include <hip/hip_bf16.h>
#include <math.h>

// Problem constants (match reference)
#define NN 50000
#define EE 800000
#define DD 256
#define HH 8
#define CC 32
#define DEdim 64
#define NEG_SLOPE 0.2f
#define LN_EPS 1e-5f

typedef __attribute__((ext_vector_type(16))) __bf16 bf16x16;
typedef __attribute__((ext_vector_type(8)))  __bf16 bf16x8;
typedef __attribute__((ext_vector_type(8)))  float  f32x8;

// ---------------------------------------------------------------------------
// WMMA 16x16x32 bf16 operand slice, converting f32 -> bf16 in registers.
// Lane L supplies row/col (L%16); k0 = kc*32 + (L/16)*8.
// elems 0..7  <- [k0 .. k0+7],  elems 8..15 <- [k0+16 .. k0+23]
// ---------------------------------------------------------------------------
__device__ __forceinline__ bf16x16 load_tile_f32cvt(const float* __restrict__ p,
                                                    int stride, long row, int k0) {
  const float* r = p + row * (long)stride + k0;
  float4 a = *reinterpret_cast<const float4*>(r);
  float4 b = *reinterpret_cast<const float4*>(r + 4);
  float4 c = *reinterpret_cast<const float4*>(r + 16);
  float4 d = *reinterpret_cast<const float4*>(r + 20);
  bf16x16 v;
  v[0]  = (__bf16)a.x; v[1]  = (__bf16)a.y; v[2]  = (__bf16)a.z; v[3]  = (__bf16)a.w;
  v[4]  = (__bf16)b.x; v[5]  = (__bf16)b.y; v[6]  = (__bf16)b.z; v[7]  = (__bf16)b.w;
  v[8]  = (__bf16)c.x; v[9]  = (__bf16)c.y; v[10] = (__bf16)c.z; v[11] = (__bf16)c.w;
  v[12] = (__bf16)d.x; v[13] = (__bf16)d.y; v[14] = (__bf16)d.z; v[15] = (__bf16)d.w;
  return v;
}

// Same slice from a pre-converted bf16 matrix: 2x16B loads, zero cvt.
__device__ __forceinline__ bf16x16 load_tile_bf16(const __bf16* __restrict__ p,
                                                  int stride, long row, int k0) {
  const __bf16* r = p + row * (long)stride + k0;
  bf16x8 lo = *reinterpret_cast<const bf16x8*>(r);
  bf16x8 hi = *reinterpret_cast<const bf16x8*>(r + 16);
  bf16x16 v;
#pragma unroll
  for (int i = 0; i < 8; ++i) { v[i] = lo[i]; v[8 + i] = hi[i]; }
  return v;
}

__device__ __forceinline__ f32x8 wmma_bf16(bf16x16 a, bf16x16 b, f32x8 c) {
  // (neg_a, A, neg_b, B, c_mod, C, reuse_a, reuse_b)
  return __builtin_amdgcn_wmma_f32_16x16x32_bf16(false, a, false, b, (short)0, c,
                                                 false, false);
}

// float atomic max via signed/unsigned int trick (valid with -inf init)
__device__ __forceinline__ void atomicMaxF(float* addr, float value) {
  if (value >= 0.0f)
    atomicMax((int*)addr, __float_as_int(value));
  else
    atomicMin((unsigned int*)addr, __float_as_uint(value));
}

// ---------------------------------------------------------------------------
// Kernel 0a: initialize workspace (agg = 0, denom = 0, amax = -inf)
// ---------------------------------------------------------------------------
__global__ __launch_bounds__(256) void init_ws(float* __restrict__ agg,
                                               float* __restrict__ amax,
                                               float* __restrict__ denom) {
  long i      = (long)blockIdx.x * blockDim.x + threadIdx.x;
  long stride = (long)gridDim.x * blockDim.x;
  for (long t = i; t < (long)NN * DD; t += stride) agg[t] = 0.0f;
  for (long t = i; t < (long)NN * HH; t += stride) {
    amax[t]  = __uint_as_float(0xFF800000u);  // -inf
    denom[t] = 0.0f;
  }
}

// ---------------------------------------------------------------------------
// Kernel 0b: one-shot weight conversion f32 -> bf16 (weights are re-read by
// thousands of blocks; convert once so hot loops load bf16 directly).
// ---------------------------------------------------------------------------
__global__ __launch_bounds__(256) void cvt_weights(
    const float* __restrict__ Wl, const float* __restrict__ Wr,
    const float* __restrict__ Wres, const float* __restrict__ We,
    __bf16* __restrict__ wl, __bf16* __restrict__ wr,
    __bf16* __restrict__ wres, __bf16* __restrict__ we) {
  long i      = (long)blockIdx.x * blockDim.x + threadIdx.x;
  long stride = (long)gridDim.x * blockDim.x;
  for (long t = i; t < (long)DD * DD; t += stride) {
    wl[t]   = (__bf16)Wl[t];
    wr[t]   = (__bf16)Wr[t];
    wres[t] = (__bf16)Wres[t];
  }
  for (long t = i; t < (long)DD * DEdim; t += stride) we[t] = (__bf16)We[t];
}

// ---------------------------------------------------------------------------
// Kernel 1: node GEMMs.  Per 16-node tile, compute [xl | xr | xres] (768 cols
// = 48 WMMA column tiles).  8 waves/block, wave w owns tiles {w, w+8, ...}.
// A (16x256, 8 K-chunks) converted once and reused across all column tiles;
// B tiles load pre-converted bf16 weights.
// ---------------------------------------------------------------------------
__global__ __launch_bounds__(256) void node_gemm(
    const float* __restrict__ x,
    const __bf16* __restrict__ Wl, const float* __restrict__ bl,
    const __bf16* __restrict__ Wr, const float* __restrict__ br,
    const __bf16* __restrict__ Wres,
    float* __restrict__ xl, float* __restrict__ xr, float* __restrict__ xres) {
  const int wave = threadIdx.x >> 5;
  const int lane = threadIdx.x & 31;
  const int half = lane >> 4;
  const int lrow = lane & 15;
  const long r0  = (long)blockIdx.x * 16;

  bf16x16 A[8];
#pragma unroll
  for (int kc = 0; kc < 8; ++kc)
    A[kc] = load_tile_f32cvt(x, DD, r0 + lrow, kc * 32 + half * 8);

#pragma unroll
  for (int tt = 0; tt < 6; ++tt) {
    const int tile = wave + tt * 8;  // 0..47
    const int ct   = tile * 16;
    const __bf16* W;
    const float* bp;
    float* dstp;
    int col;
    if (ct < 256)      { W = Wl;   bp = bl;      dstp = xl;   col = ct;       }
    else if (ct < 512) { W = Wr;   bp = br;      dstp = xr;   col = ct - 256; }
    else               { W = Wres; bp = nullptr; dstp = xres; col = ct - 512; }

    f32x8 acc;
    const float bv = bp ? bp[col + lrow] : 0.0f;
#pragma unroll
    for (int i = 0; i < 8; ++i) acc[i] = bv;

#pragma unroll
    for (int kc = 0; kc < 8; ++kc) {
      bf16x16 B = load_tile_bf16(W, DD, col + lrow, kc * 32 + half * 8);
      acc = wmma_bf16(A[kc], B, acc);
    }
#pragma unroll
    for (int i = 0; i < 8; ++i) {
      const int m = i + 8 * half;
      dstp[(r0 + m) * DD + col + lrow] = acc[i];
    }
  }
}

// ---------------------------------------------------------------------------
// Kernel 2: fused edge-embedding GEMM + attention logits.
// One block per 16-edge tile; wave = head h (owns 32 channels = 2 col tiles).
// ee computed with 4 WMMAs; never written to memory.
// ---------------------------------------------------------------------------
__global__ __launch_bounds__(256) void edge_logits(
    const float* __restrict__ ea, const __bf16* __restrict__ We,
    const float* __restrict__ att, const int* __restrict__ eidx,
    const float* __restrict__ xl, const float* __restrict__ xr,
    float* __restrict__ logit, float* __restrict__ amax) {
  const int h    = threadIdx.x >> 5;  // head
  const int lane = threadIdx.x & 31;
  const int half = lane >> 4;
  const int lrow = lane & 15;
  const long e0  = (long)blockIdx.x * 16;
  const int  n0  = h * CC;

  // A: edge_attr tile 16x64 (two K=32 chunks), f32 -> bf16 in registers
  bf16x16 A0 = load_tile_f32cvt(ea, DEdim, e0 + lrow, 0  + half * 8);
  bf16x16 A1 = load_tile_f32cvt(ea, DEdim, e0 + lrow, 32 + half * 8);

  f32x8 acc0 = {0.f,0.f,0.f,0.f,0.f,0.f,0.f,0.f};
  f32x8 acc1 = {0.f,0.f,0.f,0.f,0.f,0.f,0.f,0.f};
  bf16x16 B;
  B = load_tile_bf16(We, DEdim, n0 + lrow,      0  + half * 8); acc0 = wmma_bf16(A0, B, acc0);
  B = load_tile_bf16(We, DEdim, n0 + lrow,      32 + half * 8); acc0 = wmma_bf16(A1, B, acc0);
  B = load_tile_bf16(We, DEdim, n0 + 16 + lrow, 0  + half * 8); acc1 = wmma_bf16(A0, B, acc1);
  B = load_tile_bf16(We, DEdim, n0 + 16 + lrow, 32 + half * 8); acc1 = wmma_bf16(A1, B, acc1);

  const float attA = att[n0 + lrow];
  const float attB = att[n0 + 16 + lrow];
  const int* __restrict__ srcp = eidx;
  const int* __restrict__ dstp = eidx + EE;

#pragma unroll
  for (int i = 0; i < 8; ++i) {
    const int  m = i + 8 * half;
    const long e = e0 + m;
    const long s = srcp[e];
    const long d = dstp[e];
    float b0 = xl[s * DD + n0 + lrow]      + xr[d * DD + n0 + lrow]      + acc0[i];
    float b1 = xl[s * DD + n0 + 16 + lrow] + xr[d * DD + n0 + 16 + lrow] + acc1[i];
    float g0 = b0 > 0.0f ? b0 : NEG_SLOPE * b0;
    float g1 = b1 > 0.0f ? b1 : NEG_SLOPE * b1;
    float t  = g0 * attA + g1 * attB;
    // reduce the 32 channels: butterfly within each 16-lane half
    t += __shfl_xor(t, 1, 16);
    t += __shfl_xor(t, 2, 16);
    t += __shfl_xor(t, 4, 16);
    t += __shfl_xor(t, 8, 16);
    if (lrow == i) {  // lane i (rows 0..7) and lane 16+i (rows 8..15)
      logit[e * HH + h] = t;
      atomicMaxF(&amax[d * HH + h], t);
    }
  }
}

// ---------------------------------------------------------------------------
// Kernel 3: ealpha = exp(logit - amax[dst]); denom += ealpha
// ---------------------------------------------------------------------------
__global__ __launch_bounds__(256) void edge_exp(const int* __restrict__ eidx,
                                                const float* __restrict__ amax,
                                                float* __restrict__ logit,
                                                float* __restrict__ denom) {
  const long i = (long)blockIdx.x * blockDim.x + threadIdx.x;
  if (i >= (long)EE * HH) return;
  const long e = i >> 3;
  const int  h = (int)(i & 7);
  const long d = eidx[EE + e];
  const float v = __expf(logit[i] - amax[d * HH + h]);
  logit[i] = v;
  atomicAdd(&denom[d * HH + h], v);
}

// ---------------------------------------------------------------------------
// Kernel 4: scatter ealpha-weighted messages.  One block (256 thr) per edge.
// ---------------------------------------------------------------------------
__global__ __launch_bounds__(256) void edge_scatter(const int* __restrict__ eidx,
                                                    const float* __restrict__ xl,
                                                    const float* __restrict__ ealpha,
                                                    float* __restrict__ agg) {
  const long e = blockIdx.x;
  const int  n = threadIdx.x;
  const long s = eidx[e];
  const long d = eidx[EE + e];
  const float a = ealpha[e * HH + (n >> 5)];
  atomicAdd(&agg[d * DD + n], xl[s * DD + n] * a);
}

// ---------------------------------------------------------------------------
// Kernel 5: normalize, residual, bias, LayerNorm.  One wave per node.
// Lane l owns channels {l, l+32, ..., l+224}; channel l+32j belongs to head j.
// ---------------------------------------------------------------------------
__global__ __launch_bounds__(256) void finalize(
    const float* __restrict__ agg, const float* __restrict__ denom,
    const float* __restrict__ xres, const float* __restrict__ bias,
    const float* __restrict__ gamma, const float* __restrict__ beta,
    float* __restrict__ out) {
  const int wave = threadIdx.x >> 5;
  const int lane = threadIdx.x & 31;
  const long n   = (long)blockIdx.x * 8 + wave;
  if (n >= NN) return;

  float v[8];
  float sum = 0.0f, sumsq = 0.0f;
#pragma unroll
  for (int j = 0; j < 8; ++j) {
    const int ch = lane + 32 * j;
    float dn = denom[n * HH + j];
    dn = dn > 0.0f ? dn : 1.0f;
    float val = agg[n * DD + ch] / dn + xres[n * DD + ch] + bias[ch];
    v[j] = val;
    sum += val;
    sumsq += val * val;
  }
#pragma unroll
  for (int o = 1; o < 32; o <<= 1) {
    sum   += __shfl_xor(sum, o, 32);
    sumsq += __shfl_xor(sumsq, o, 32);
  }
  const float mu  = sum * (1.0f / DD);
  const float var = sumsq * (1.0f / DD) - mu * mu;
  const float rs  = rsqrtf(var + LN_EPS);
#pragma unroll
  for (int j = 0; j < 8; ++j) {
    const int ch = lane + 32 * j;
    out[n * DD + ch] = (v[j] - mu) * rs * gamma[ch] + beta[ch];
  }
}

// ---------------------------------------------------------------------------
extern "C" void kernel_launch(void* const* d_in, const int* in_sizes, int n_in,
                              void* d_out, int out_size, void* d_ws, size_t ws_size,
                              hipStream_t stream) {
  const float* x         = (const float*)d_in[0];
  const float* edge_attr = (const float*)d_in[1];
  const int*   edge_idx  = (const int*)  d_in[2];
  const float* W_l       = (const float*)d_in[3];
  const float* b_l       = (const float*)d_in[4];
  const float* W_r       = (const float*)d_in[5];
  const float* b_r       = (const float*)d_in[6];
  const float* W_e       = (const float*)d_in[7];
  const float* att       = (const float*)d_in[8];
  const float* W_res     = (const float*)d_in[9];
  const float* bias      = (const float*)d_in[10];
  const float* gamma     = (const float*)d_in[11];
  const float* beta      = (const float*)d_in[12];
  float* out = (float*)d_out;

  float* ws = (float*)d_ws;
  const long ND = (long)NN * DD;
  float* xl    = ws;
  float* xr    = ws + ND;
  float* xres  = ws + 2 * ND;
  float* agg   = ws + 3 * ND;
  float* amax  = ws + 4 * ND;
  float* denom = amax + (long)NN * HH;
  float* elog  = denom + (long)NN * HH;       // E*H logits -> ealpha (in place)
  // bf16 weight cache after the f32 regions (all offsets 16B-aligned)
  __bf16* wbf   = (__bf16*)(elog + (long)EE * HH);
  __bf16* wl_bf   = wbf;
  __bf16* wr_bf   = wbf + (long)DD * DD;
  __bf16* wres_bf = wbf + 2 * (long)DD * DD;
  __bf16* we_bf   = wbf + 3 * (long)DD * DD;

  init_ws<<<12800, 256, 0, stream>>>(agg, amax, denom);
  cvt_weights<<<256, 256, 0, stream>>>(W_l, W_r, W_res, W_e,
                                       wl_bf, wr_bf, wres_bf, we_bf);
  node_gemm<<<NN / 16, 256, 0, stream>>>(x, wl_bf, b_l, wr_bf, b_r, wres_bf,
                                         xl, xr, xres);
  edge_logits<<<EE / 16, 256, 0, stream>>>(edge_attr, we_bf, att, edge_idx, xl, xr,
                                           elog, amax);
  edge_exp<<<(int)(((long)EE * HH + 255) / 256), 256, 0, stream>>>(edge_idx, amax,
                                                                   elog, denom);
  edge_scatter<<<EE, 256, 0, stream>>>(edge_idx, xl, elog, agg);
  finalize<<<NN / 8, 256, 0, stream>>>(agg, denom, xres, bias, gamma, beta, out);
}
